// MLPBaseline_87273735454925
// MI455X (gfx1250) — compile-verified, matrix operand order checked by reading, not run on hardware
//
#include <hip/hip_runtime.h>
#include <hip/hip_bf16.h>

// ---------------------------------------------------------------------------
// CDNA5 (gfx1250): bf16 WMMA GEMMs (f32 accum) + TDM (tensor_load_to_lds)
// staging of weight panels into LDS with double buffering.
// Pipeline: convert/pad -> GEMM1 -> BN stats -> BN+GELU -> GEMM2 -> BN
//           -> edge WMMA scorer -> decoder GEMMs (GELU fused) -> classifier.
// ---------------------------------------------------------------------------

typedef __bf16 bf16;
typedef __attribute__((ext_vector_type(16))) __bf16 v16bf;
typedef __attribute__((ext_vector_type(8)))  __bf16 v8bf;
typedef __attribute__((ext_vector_type(8)))  float  v8f;
typedef __attribute__((ext_vector_type(4)))  unsigned int u32x4;
typedef __attribute__((ext_vector_type(4)))  int i32x4;
typedef __attribute__((ext_vector_type(8)))  int i32x8;

#if defined(__has_builtin)
#if __has_builtin(__builtin_amdgcn_tensor_load_to_lds) && \
    __has_builtin(__builtin_amdgcn_s_wait_tensorcnt)
#define USE_TDM 1
#endif
#endif
#ifndef USE_TDM
#define USE_TDM 0
#endif

union ABFrag { v16bf v; v8bf h[2]; };

__device__ __forceinline__ float gelu_exact(float x) {
    // torch nn.GELU default (exact erf)
    return 0.5f * x * (1.0f + erff(x * 0.70710678118654752440f));
}

__device__ __forceinline__ unsigned lds_off_of(const void* p) {
    // LDS aperture: flat address [31:0] is the LDS offset (ISA 10.2)
    return (unsigned)(size_t)p;
}

#if USE_TDM
// Issue a TDM 2D tile load: 256 rows x 64 cols of bf16 from Bt (row stride =
// row_stride_elems) into LDS at lds_addr. Rows >= rows_valid zero-fill (OOB).
// D# bitfields per CDNA5 ISA.
__device__ __forceinline__ void tdm_issue_b_chunk(const bf16* gptr, unsigned lds_addr,
                                                  int rows_valid, int row_stride_elems) {
    unsigned long long ga = (unsigned long long)(size_t)gptr;
    u32x4 g0;
    g0[0] = 1u;                                          // count = 1 (valid user D#)
    g0[1] = lds_addr;                                    // lds_addr [63:32]
    g0[2] = (unsigned)(ga & 0xffffffffu);                // global_addr [95:64]
    g0[3] = (unsigned)((ga >> 32) & 0x1ffffffu)          // global_addr [120:96]
          | (2u << 30);                                  // type = 2 ("image")
    unsigned td0 = (unsigned)row_stride_elems;           // tensor_dim0 (row length)
    unsigned td1 = (unsigned)rows_valid;                 // tensor_dim1 (rows in range)
    i32x8 g1;
    g1[0] = (int)(1u << 16);                             // wg_mask=0, data_size=1 (2B)
    g1[1] = (int)((td0 & 0xffffu) << 16);                // tensor_dim0[15:0] @ 63:48
    g1[2] = (int)(((td0 >> 16) & 0xffffu)                // tensor_dim0[31:16]
                | ((td1 & 0xffffu) << 16));              // tensor_dim1[15:0] @ 95:80
    g1[3] = (int)(((td1 >> 16) & 0xffffu)                // tensor_dim1[31:16]
                | (64u << 16));                          // tile_dim0 = 64 @ 127:112
    g1[4] = (int)256u;                                   // tile_dim1 = 256; tile_dim2 = 0
    g1[5] = (int)td0;                                    // tensor_dim0_stride[31:0]
    g1[6] = 0;                                           // stride[47:32]=0, dim1_stride=0
    g1[7] = 0;
    i32x4 z4 = {0, 0, 0, 0};
#if defined(__clang_major__) && (__clang_major__ >= 23)
    i32x8 z8 = {0, 0, 0, 0, 0, 0, 0, 0};
    __builtin_amdgcn_tensor_load_to_lds(g0, g1, z4, z4, z8, 0);
#else
    __builtin_amdgcn_tensor_load_to_lds(g0, g1, z4, z4, 0);
#endif
}
#endif // USE_TDM

// ---------------- conversion / layout kernels ----------------
// src [rows,K] fp32 -> dst [rows,KP] bf16, zero-padding cols K..KP-1
__global__ void pad_convert_rows_kernel(const float* __restrict__ src, bf16* __restrict__ dst,
                                        long rows, int K, int KP) {
    long total = rows * KP;
    long i = (long)blockIdx.x * blockDim.x + threadIdx.x;
    long stride = (long)gridDim.x * blockDim.x;
    for (; i < total; i += stride) {
        long r = i / KP;
        int  c = (int)(i - r * KP);
        dst[i] = (c < K) ? (bf16)src[r * K + c] : (bf16)0.0f;
    }
}

// src [K,N] fp32 -> dst [N,KP] bf16 (transpose), zero-padding k in [K,KP)
__global__ void transpose_pad_f32_to_bf16_kernel(const float* __restrict__ src,
                                                 bf16* __restrict__ dst,
                                                 int K, int N, int KP) {
    int n = blockIdx.x * blockDim.x + threadIdx.x;
    int k = blockIdx.y * blockDim.y + threadIdx.y;
    if (n < N && k < KP)
        dst[(long)n * KP + k] = (k < K) ? (bf16)src[(long)k * N + n] : (bf16)0.0f;
}

__global__ void zero_f32_kernel(float* __restrict__ p, int n) {
    int i = blockIdx.x * blockDim.x + threadIdx.x;
    if (i < n) p[i] = 0.0f;
}

// ---------------- WMMA GEMM: C[M,N] = A[M,K] @ Bt[N,K]^T + bias ----------------
// Workgroup = 8 waves as 2(M) x 4(N); wave tile 16M x 64N (4 accumulators);
// block tile 32M x 256N. B panel (256 rows x 64 K, 32 KB) staged in LDS by the
// Tensor Data Mover, double buffered; A fragments loaded direct from global.
// K must be a multiple of 64 (padded). EPILOGUE 0: fp32 C. 1: bf16 gelu -> Cb.
template<int EPILOGUE>
__global__ __launch_bounds__(256) void gemm_wmma_bf16_kernel(
    const bf16* __restrict__ A, const bf16* __restrict__ Bt,
    const float* __restrict__ bias,
    float* __restrict__ C, bf16* __restrict__ Cb,
    int K, int N)
{
    __shared__ bf16 ldsB[2][256 * 64];   // 2 x 32 KB double buffer
    const int wave = threadIdx.x >> 5;
    const int lane = threadIdx.x & 31;
    const int mw   = (wave >> 2) << 4;          // 0 / 16
    const int nw   = (wave & 3) << 6;           // 0 / 64 / 128 / 192
    const int m0   = (blockIdx.y << 5) + mw;
    const int nblock = blockIdx.x << 8;
    const int row   = lane & 15;
    const int khalf = (lane >> 4) << 3;         // CDNA5 16-bit fragment K-half select
    const int col   = lane & 15;
    const bf16* arow = A + (long)(m0 + row) * K;
    const int nchunks = K >> 6;
    const int rows_valid = N - nblock;

    v8f acc[4] = {};

#if USE_TDM
    if (wave == 0)
        tdm_issue_b_chunk(Bt + (long)nblock * K, lds_off_of(&ldsB[0][0]), rows_valid, K);
#endif
    for (int c = 0; c < nchunks; ++c) {
        const int kc = c << 6;
#if USE_TDM
        if (wave == 0) {
            if (c + 1 < nchunks) {
                tdm_issue_b_chunk(Bt + (long)nblock * K + (kc + 64),
                                  lds_off_of(&ldsB[(c + 1) & 1][0]), rows_valid, K);
                __builtin_amdgcn_s_wait_tensorcnt(1);   // chunk c landed (in-order TDM)
            } else {
                __builtin_amdgcn_s_wait_tensorcnt(0);
            }
        }
        __syncthreads();
        const bf16* bbuf = &ldsB[c & 1][0];
#else
        // cooperative staging fallback: 256 threads copy 256x64 bf16 (32 KB)
        {
            const bf16* gsrc = Bt + (long)nblock * K + kc;
            #pragma unroll
            for (int j = 0; j < 8; ++j) {
                int u = (int)threadIdx.x + (j << 8);    // 16B unit id, 2048 units
                int r = u >> 3;
                int o = (u & 7) << 3;
                v8bf v = {};
                if (r < rows_valid) v = *(const v8bf*)(gsrc + (long)r * K + o);
                *(v8bf*)(&ldsB[0][r * 64 + o]) = v;
            }
        }
        __syncthreads();
        const bf16* bbuf = &ldsB[0][0];
#endif
        #pragma unroll
        for (int kk = 0; kk < 64; kk += 32) {
            ABFrag a;
            a.h[0] = *(const v8bf*)(arow + kc + kk + khalf);
            a.h[1] = *(const v8bf*)(arow + kc + kk + 16 + khalf);
            __builtin_prefetch(arow + kc + 128, 0, 3);   // global_prefetch A chunk c+2
            #pragma unroll
            for (int t = 0; t < 4; ++t) {
                const bf16* brow = bbuf + (nw + (t << 4) + row) * 64;
                ABFrag b;
                b.h[0] = *(const v8bf*)(brow + kk + khalf);
                b.h[1] = *(const v8bf*)(brow + kk + 16 + khalf);
                acc[t] = __builtin_amdgcn_wmma_f32_16x16x32_bf16(
                    false, a.v, false, b.v, (short)0, acc[t], false, false);
            }
        }
        __syncthreads();   // all waves done reading before TDM overwrites this buffer
    }
    // C/D layout: VGPR i, lanes 0-15 -> M=i; lanes 16-31 -> M=8+i; N = lane&15
    const int rbase = (lane >> 4) << 3;
    #pragma unroll
    for (int t = 0; t < 4; ++t) {
        const int n0 = nblock + nw + (t << 4);
        if (n0 >= N) continue;                 // N=2000 tail (wave-uniform)
        const float bv = bias[n0 + col];
        #pragma unroll
        for (int i = 0; i < 8; i++) {
            float v = acc[t][i] + bv;
            long idx = (long)(m0 + rbase + i) * N + n0 + col;
            if (EPILOGUE == 0) C[idx]  = v;
            else               Cb[idx] = (bf16)gelu_exact(v);
        }
    }
}

// ---------------- BatchNorm column stats (biased, training-mode) ----------------
__global__ void col_stats_kernel(const float* __restrict__ t, int M, int N, int rows_per_blk,
                                 float* __restrict__ sum, float* __restrict__ sumsq)
{
    int c = blockIdx.x * blockDim.x + threadIdx.x;
    if (c >= N) return;
    int r0 = blockIdx.y * rows_per_blk;
    int r1 = r0 + rows_per_blk; if (r1 > M) r1 = M;
    float s = 0.0f, s2 = 0.0f;
    for (int r = r0; r < r1; r++) {
        float v = t[(long)r * N + c];
        s += v; s2 += v * v;
    }
    atomicAdd(&sum[c], s);
    atomicAdd(&sumsq[c], s2);
}

__global__ void bn_gelu_kernel(const float* __restrict__ t,
                               const float* __restrict__ sum, const float* __restrict__ sumsq,
                               const float* __restrict__ gamma, const float* __restrict__ beta,
                               int M, int N,
                               float* __restrict__ out_f32, bf16* __restrict__ out_bf16)
{
    long n_total = (long)M * N;
    long i = (long)blockIdx.x * blockDim.x + threadIdx.x;
    long stride = (long)gridDim.x * blockDim.x;
    float invM = 1.0f / (float)M;
    for (; i < n_total; i += stride) {
        int c = (int)(i % N);
        float mean = sum[c] * invM;
        float var  = sumsq[c] * invM - mean * mean;
        float xn = (t[i] - mean) * rsqrtf(var + 1e-5f) * gamma[c] + beta[c];
        float g = gelu_exact(xn);
        if (out_f32) out_f32[i] = g;
        out_bf16[i] = (bf16)g;
    }
}

// ---------------- Edge scorer: gathered-A WMMA GEMM + fused GELU/dot ----------------
// One workgroup = 32 edges (2 M tiles). 8 waves; wave w covers cols [32w, 32w+32).
// Each wave: 2x2 tiles -> 4 WMMAs per K step; B fragments amortized over 2 M tiles.
// A row e = concat(h[src[e]], h[dst[e]]) over K=512; boundary 256 aligns with K step.
__global__ __launch_bounds__(256) void edge_score_kernel(
    const bf16* __restrict__ Hb,          // [N_NODES, 256] bf16
    const int* __restrict__ srcIdx, const int* __restrict__ dstIdx,
    const bf16* __restrict__ Wl1t,        // [256, 512] bf16 (transposed)
    const float* __restrict__ bl1,        // [256]
    const float* __restrict__ Wl2,        // [256]
    const float* __restrict__ bl2,        // [1]
    float* __restrict__ out)
{
    __shared__ float s_acc[32];
    const int wave = threadIdx.x >> 5;
    const int lane = threadIdx.x & 31;
    const int e0 = blockIdx.x << 5;
    const int row   = lane & 15;
    const int khalf = (lane >> 4) << 3;
    const bf16* hs0 = Hb + (long)srcIdx[e0 + row] * 256;
    const bf16* ht0 = Hb + (long)dstIdx[e0 + row] * 256;
    const bf16* hs1 = Hb + (long)srcIdx[e0 + 16 + row] * 256;
    const bf16* ht1 = Hb + (long)dstIdx[e0 + 16 + row] * 256;
    if (threadIdx.x < 32) s_acc[threadIdx.x] = 0.0f;
    __syncthreads();

    const int n0 = wave << 5;
    const bf16* b0row = Wl1t + (long)(n0 + row) * 512;
    const bf16* b1row = Wl1t + (long)(n0 + 16 + row) * 512;
    v8f acc00 = {}, acc01 = {}, acc10 = {}, acc11 = {};
    for (int k = 0; k < 512; k += 32) {
        const bf16* f0 = (k < 256) ? (hs0 + k) : (ht0 + (k - 256));
        const bf16* f1 = (k < 256) ? (hs1 + k) : (ht1 + (k - 256));
        ABFrag a0, a1, b0, b1;
        a0.h[0] = *(const v8bf*)(f0 + khalf);
        a0.h[1] = *(const v8bf*)(f0 + 16 + khalf);
        a1.h[0] = *(const v8bf*)(f1 + khalf);
        a1.h[1] = *(const v8bf*)(f1 + 16 + khalf);
        b0.h[0] = *(const v8bf*)(b0row + k + khalf);
        b0.h[1] = *(const v8bf*)(b0row + k + 16 + khalf);
        b1.h[0] = *(const v8bf*)(b1row + k + khalf);
        b1.h[1] = *(const v8bf*)(b1row + k + 16 + khalf);
        acc00 = __builtin_amdgcn_wmma_f32_16x16x32_bf16(false, a0.v, false, b0.v, (short)0, acc00, false, false);
        acc01 = __builtin_amdgcn_wmma_f32_16x16x32_bf16(false, a0.v, false, b1.v, (short)0, acc01, false, false);
        acc10 = __builtin_amdgcn_wmma_f32_16x16x32_bf16(false, a1.v, false, b0.v, (short)0, acc10, false, false);
        acc11 = __builtin_amdgcn_wmma_f32_16x16x32_bf16(false, a1.v, false, b1.v, (short)0, acc11, false, false);
    }
    const int col = lane & 15;
    const float w2a = Wl2[n0 + col];
    const float w2b = Wl2[n0 + 16 + col];
    const float ba  = bl1[n0 + col];
    const float bb  = bl1[n0 + 16 + col];
    float p0[8], p1[8];
    #pragma unroll
    for (int i = 0; i < 8; i++) {
        p0[i] = gelu_exact(acc00[i] + ba) * w2a + gelu_exact(acc01[i] + bb) * w2b;
        p1[i] = gelu_exact(acc10[i] + ba) * w2a + gelu_exact(acc11[i] + bb) * w2b;
    }
    // reduce over the 16 lanes that share an output edge row (different N columns)
    #pragma unroll
    for (int off = 8; off >= 1; off >>= 1) {
        #pragma unroll
        for (int i = 0; i < 8; i++) {
            p0[i] += __shfl_xor(p0[i], off, 16);
            p1[i] += __shfl_xor(p1[i], off, 16);
        }
    }
    if (col == 0) {
        const int rbase = (lane >> 4) << 3;
        #pragma unroll
        for (int i = 0; i < 8; i++) {
            atomicAdd(&s_acc[rbase + i], p0[i]);        // ds_add_f32
            atomicAdd(&s_acc[16 + rbase + i], p1[i]);
        }
    }
    __syncthreads();
    if (threadIdx.x < 32) out[e0 + threadIdx.x] = s_acc[threadIdx.x] + bl2[0];
}

// ---------------- tiny classifier (N=10, too thin for WMMA) ----------------
__global__ void classifier_kernel(const float* __restrict__ h, const float* __restrict__ Wc,
                                  const float* __restrict__ bc, float* __restrict__ out, int n_nodes)
{
    int idx = blockIdx.x * blockDim.x + threadIdx.x;
    if (idx >= n_nodes * 10) return;
    int node = idx / 10, c = idx % 10;
    const float* hr = h + (long)node * 256;
    float s = bc[c];
    #pragma unroll 8
    for (int k = 0; k < 256; k++) s += hr[k] * Wc[k * 10 + c];
    out[idx] = s;
}

// ---------------------------------------------------------------------------
extern "C" void kernel_launch(void* const* d_in, const int* in_sizes, int n_in,
                              void* d_out, int out_size, void* d_ws, size_t ws_size,
                              hipStream_t stream)
{
    constexpr int NN = 20000, NG = 2000, NGP = 2048, H1 = 512, H = 256, NT = 10, NE = 320000;

    const float* x   = (const float*)d_in[0];
    const int*   ei  = (const int*)d_in[1];      // [2, NE]
    const float* W1  = (const float*)d_in[2];
    const float* b1  = (const float*)d_in[3];
    const float* g1  = (const float*)d_in[4];
    const float* be1 = (const float*)d_in[5];
    const float* W2  = (const float*)d_in[6];
    const float* b2  = (const float*)d_in[7];
    const float* g2  = (const float*)d_in[8];
    const float* be2 = (const float*)d_in[9];
    const float* Wl1 = (const float*)d_in[10];
    const float* bl1 = (const float*)d_in[11];
    const float* Wl2 = (const float*)d_in[12];
    const float* bl2 = (const float*)d_in[13];
    const float* Wd1 = (const float*)d_in[14];
    const float* bd1 = (const float*)d_in[15];
    const float* Wd2 = (const float*)d_in[16];
    const float* bd2 = (const float*)d_in[17];
    const float* Wc  = (const float*)d_in[18];
    const float* bc  = (const float*)d_in[19];

    // output layout: [lr_scores | reconstruction | cell_type | h]
    float* lr    = (float*)d_out;
    float* recon = lr + NE;
    float* cell  = recon + (long)NN * NG;
    float* hout  = cell + (long)NN * NT;

    // workspace carve-out (256B aligned)
    char* wptr = (char*)d_ws;
    auto alloc = [&](size_t bytes) -> void* {
        void* p = (void*)wptr;
        wptr += (bytes + 255) & ~(size_t)255;
        return p;
    };
    bf16*  x_bf  = (bf16*)alloc((size_t)NN * NGP * 2);  // [20000,2048] K-padded
    bf16*  W1t   = (bf16*)alloc((size_t)H1 * NGP * 2);  // [512,2048]  K-padded
    bf16*  W2t   = (bf16*)alloc((size_t)H  * H1 * 2);   // [256,512]
    bf16*  Wl1t  = (bf16*)alloc((size_t)H  * (2*H) * 2);// [256,512]
    bf16*  Wd1t  = (bf16*)alloc((size_t)H1 * H  * 2);   // [512,256]
    bf16*  Wd2t  = (bf16*)alloc((size_t)NG * H1 * 2);   // [2000,512]
    float* t1    = (float*)alloc((size_t)NN * H1 * 4);
    bf16*  h1_bf = (bf16*)alloc((size_t)NN * H1 * 2);
    float* t2    = (float*)alloc((size_t)NN * H  * 4);
    bf16*  h_bf  = (bf16*)alloc((size_t)NN * H  * 2);
    bf16*  d1_bf = (bf16*)alloc((size_t)NN * H1 * 2);
    float* sum1  = (float*)alloc(H1 * 4);
    float* sq1   = (float*)alloc(H1 * 4);
    float* sum2  = (float*)alloc(H * 4);
    float* sq2   = (float*)alloc(H * 4);

    // 1) precision conversion (+K padding) and weight transposes
    pad_convert_rows_kernel<<<4096, 256, 0, stream>>>(x, x_bf, NN, NG, NGP);
    dim3 tb(32, 8);
    transpose_pad_f32_to_bf16_kernel<<<dim3((H1 + 31) / 32, (NGP + 7) / 8), tb, 0, stream>>>(W1, W1t, NG, H1, NGP);
    transpose_pad_f32_to_bf16_kernel<<<dim3((H  + 31) / 32, (H1 + 7) / 8), tb, 0, stream>>>(W2, W2t, H1, H, H1);
    transpose_pad_f32_to_bf16_kernel<<<dim3((H  + 31) / 32, (H1 + 7) / 8), tb, 0, stream>>>(Wl1, Wl1t, 2 * H, H, 2 * H);
    transpose_pad_f32_to_bf16_kernel<<<dim3((H1 + 31) / 32, (H  + 7) / 8), tb, 0, stream>>>(Wd1, Wd1t, H, H1, H);
    transpose_pad_f32_to_bf16_kernel<<<dim3((NG + 31) / 32, (H1 + 7) / 8), tb, 0, stream>>>(Wd2, Wd2t, H1, NG, H1);

    // 2) encoder layer 1: t1 = x @ W1 + b1   (K padded to 2048)
    gemm_wmma_bf16_kernel<0><<<dim3(H1 / 256, NN / 32), 256, 0, stream>>>(
        x_bf, W1t, b1, t1, nullptr, NGP, H1);
    zero_f32_kernel<<<(H1 + 255) / 256, 256, 0, stream>>>(sum1, H1);
    zero_f32_kernel<<<(H1 + 255) / 256, 256, 0, stream>>>(sq1, H1);
    col_stats_kernel<<<dim3((H1 + 255) / 256, 80), 256, 0, stream>>>(t1, NN, H1, 250, sum1, sq1);
    bn_gelu_kernel<<<4096, 256, 0, stream>>>(t1, sum1, sq1, g1, be1, NN, H1, nullptr, h1_bf);

    // 3) encoder layer 2: t2 = h1 @ W2 + b2 ; h = gelu(bn(t2)) (fp32 out + bf16 copy)
    gemm_wmma_bf16_kernel<0><<<dim3(1, NN / 32), 256, 0, stream>>>(
        h1_bf, W2t, b2, t2, nullptr, H1, H);
    zero_f32_kernel<<<(H + 255) / 256, 256, 0, stream>>>(sum2, H);
    zero_f32_kernel<<<(H + 255) / 256, 256, 0, stream>>>(sq2, H);
    col_stats_kernel<<<dim3((H + 255) / 256, 80), 256, 0, stream>>>(t2, NN, H, 250, sum2, sq2);
    bn_gelu_kernel<<<4096, 256, 0, stream>>>(t2, sum2, sq2, g2, be2, NN, H, hout, h_bf);

    // 4) edge scorer (gathered-A WMMA, 32 edges / workgroup)
    edge_score_kernel<<<NE / 32, 256, 0, stream>>>(
        h_bf, ei, ei + NE, Wl1t, bl1, Wl2, bl2, lr);

    // 5) decoder: d1 = gelu(h @ Wd1 + bd1) [GELU fused, bf16 out]
    gemm_wmma_bf16_kernel<1><<<dim3(H1 / 256, NN / 32), 256, 0, stream>>>(
        h_bf, Wd1t, bd1, nullptr, d1_bf, H, H1);
    //    reconstruction = d1 @ Wd2 + bd2  (N=2000: tile-level tail handling)
    gemm_wmma_bf16_kernel<0><<<dim3((NG + 255) / 256, NN / 32), 256, 0, stream>>>(
        d1_bf, Wd2t, bd2, recon, nullptr, H1, NG);

    // 6) classifier
    classifier_kernel<<<(NN * NT + 255) / 256, 256, 0, stream>>>(hout, Wc, bc, cell, NN);

    (void)in_sizes; (void)n_in; (void)out_size; (void)ws_size;
}
